// EvoformerBlock_57561151701392
// MI455X (gfx1250) — matrix operand order repaired and checked by reading, not compiled, not run on hardware
//
#include <hip/hip_runtime.h>
#include <cmath>

// ============================================================================
// Evoformer block for gfx1250 (MI455X). All contractions via
// V_WMMA_F32_16X16X32_F16 (f16 operands, f32 accumulate). ~170 GFLOP total,
// working set ~190MB (~= L2). Workspace requirement: >= 191 MiB.
// GEMM: one wave owns a 32x32 C block (2x2 WMMA tiles), clamped unconditional
// fragment loads (no exec-mask juggling in the k-loop).
// ============================================================================

typedef __attribute__((ext_vector_type(16))) _Float16 v16h;
typedef __attribute__((ext_vector_type(8)))  _Float16 v8h;
typedef __attribute__((ext_vector_type(8)))  float    v8f;

static const int S_ = 128, N_ = 256, CM = 256, CZ = 128;
static const int HM = 8, HP = 4, CHD = 32, COPM = 32, CMUL = 128, TN = 4;

enum { FLAG_RELU = 1, FLAG_F16OUT = 2, FLAG_ACC = 4, FLAG_ROWSCALE = 8, FLAG_GATE = 16 };

// ---- input index map: jax flattens dicts in sorted-key order ---------------
enum InIdx {
  IN_M = 0, IN_Z, IN_MSA_MASK, IN_PAIR_MASK,
  CA_G_B, CA_G_W, CA_K_B, CA_K_W, CA_O_B, CA_O_W, CA_Q_B, CA_Q_W, CA_V_B, CA_V_W,
  CA_LN_BIAS, CA_LN_SCALE,
  MT_L1_B, MT_L1_W, MT_L2_B, MT_L2_W, MT_LN_BIAS, MT_LN_SCALE,
  OPM_L1_B, OPM_L1_W, OPM_L2_B, OPM_L2_W, OPM_LN_BIAS, OPM_LN_SCALE, OPM_OUT_B, OPM_OUT_W,
  PT_L1_B, PT_L1_W, PT_L2_B, PT_L2_W, PT_LN_BIAS, PT_LN_SCALE,
  RA_G_B, RA_G_W, RA_K_B, RA_K_W, RA_O_B, RA_O_W, RA_Q_B, RA_Q_W, RA_V_B, RA_V_W,
  RA_LNM_BIAS, RA_LNM_SCALE, RA_LNZ_BIAS, RA_LNZ_SCALE, RA_ZB_B, RA_ZB_W,
  TAE_G_B, TAE_G_W, TAE_K_B, TAE_K_W, TAE_O_B, TAE_O_W, TAE_Q_B, TAE_Q_W, TAE_V_B, TAE_V_W,
  TAE_LN_BIAS, TAE_LN_SCALE, TAE_TB_B, TAE_TB_W,
  TAS_G_B, TAS_G_W, TAS_K_B, TAS_K_W, TAS_O_B, TAS_O_W, TAS_Q_B, TAS_Q_W, TAS_V_B, TAS_V_W,
  TAS_LN_BIAS, TAS_LN_SCALE, TAS_TB_B, TAS_TB_W,
  TMI_AG_B, TMI_AG_W, TMI_AP_B, TMI_AP_W, TMI_BG_B, TMI_BG_W, TMI_BP_B, TMI_BP_W,
  TMI_G_B, TMI_G_W, TMI_LNI_BIAS, TMI_LNI_SCALE, TMI_LNO_BIAS, TMI_LNO_SCALE, TMI_OUT_B, TMI_OUT_W,
  TMO_AG_B, TMO_AG_W, TMO_AP_B, TMO_AP_W, TMO_BG_B, TMO_BG_W, TMO_BP_B, TMO_BP_W,
  TMO_G_B, TMO_G_W, TMO_LNI_BIAS, TMO_LNI_SCALE, TMO_LNO_BIAS, TMO_LNO_SCALE, TMO_OUT_B, TMO_OUT_W
};

// ============================================================================
// Generic strided/batched WMMA GEMM:  C[m,n] (+)= act( A[m,:] . B_t[n,:] + b )
//   A: f16, rows addressed as baseA(r) = (r/rowDivA)*rsA1 + (r%rowDivA)*rsA2,
//      K consumed in 32-chunks, chunk kc at +kc*kcsA, contiguous 32 halves.
//   B: f16 "transposed" layout: row n at n*ldb, contiguous K.
//   C: element (m,n) at batchBase + m*ldc + n*colStrideC  (f32 or f16).
// 2x2 WMMA tiles per wave; out-of-range rows/cols load clamped data (results
// discarded by the guarded store).
// ============================================================================
struct GemmArgs {
  const _Float16* A; const _Float16* B; void* C;
  const float* bias; const float* rowscale; const float* gate;
  long long sA1, sA2, sB1, sB2, sC1, sC2;          // batch strides (grid.y, grid.z)
  long long rsA1, rsA2, kcsA, ldb, ldc;
  int rowDivA, colStrideC, M, N, K, flags;
};

__global__ void __launch_bounds__(128) k_gemm(GemmArgs g) {
  const int wv = threadIdx.x >> 5, lane = threadIdx.x & 31;
  const int tilesM = (g.M + 31) >> 5, tilesN = (g.N + 31) >> 5;
  const int tile = blockIdx.x * 4 + wv;
  if (tile >= tilesM * tilesN) return;
  const int tm = tile % tilesM, tn = tile / tilesM;
  const int nl = lane & 15, grp = lane >> 4;

  const _Float16* A  = g.A + blockIdx.y * g.sA1 + blockIdx.z * g.sA2;
  const _Float16* Bp = g.B + blockIdx.y * g.sB1 + blockIdx.z * g.sB2;

  // A fragment rows (16-bit A 16x32 layout: lane(grp,nl) covers K=8g+e / 16+8g+e)
  const int am0 = tm * 32 + nl, am1 = am0 + 16;
  const int amc0 = am0 < g.M ? am0 : g.M - 1;
  const int amc1 = am1 < g.M ? am1 : g.M - 1;
  const long long ab0 = (long long)(amc0 / g.rowDivA) * g.rsA1 +
                        (long long)(amc0 % g.rowDivA) * g.rsA2;
  const long long ab1 = (long long)(amc1 / g.rowDivA) * g.rsA1 +
                        (long long)(amc1 % g.rowDivA) * g.rsA2;
  const _Float16* pa0 = A + ab0 + 8 * grp;
  const _Float16* pa1 = A + ab1 + 8 * grp;

  // B fragment cols (16-bit B 32x16 layout: lane(grp,nl) covers K=8g+e / 16+8g+e)
  const int bn0 = tn * 32 + nl, bn1 = bn0 + 16;
  const int bnc0 = bn0 < g.N ? bn0 : g.N - 1;
  const int bnc1 = bn1 < g.N ? bn1 : g.N - 1;
  const _Float16* pb0 = Bp + (long long)bnc0 * g.ldb + 8 * grp;
  const _Float16* pb1 = Bp + (long long)bnc1 * g.ldb + 8 * grp;

  v8f a00, a01, a10, a11;
  for (int i = 0; i < 8; ++i) { a00[i] = 0.f; a01[i] = 0.f; a10[i] = 0.f; a11[i] = 0.f; }

  for (int kc = g.K >> 5; kc > 0; --kc) {
    union { v8h h[2]; v16h v; } fa0, fa1, fb0, fb1;
    fa0.h[0] = *(const v8h*)pa0;       fa0.h[1] = *(const v8h*)(pa0 + 16);
    fa1.h[0] = *(const v8h*)pa1;       fa1.h[1] = *(const v8h*)(pa1 + 16);
    fb0.h[0] = *(const v8h*)pb0;       fb0.h[1] = *(const v8h*)(pb0 + 16);
    fb1.h[0] = *(const v8h*)pb1;       fb1.h[1] = *(const v8h*)(pb1 + 16);
    a00 = __builtin_amdgcn_wmma_f32_16x16x32_f16(false, fa0.v, false, fb0.v, (short)0, a00, false, false);
    a01 = __builtin_amdgcn_wmma_f32_16x16x32_f16(false, fa0.v, false, fb1.v, (short)0, a01, false, false);
    a10 = __builtin_amdgcn_wmma_f32_16x16x32_f16(false, fa1.v, false, fb0.v, (short)0, a10, false, false);
    a11 = __builtin_amdgcn_wmma_f32_16x16x32_f16(false, fa1.v, false, fb1.v, (short)0, a11, false, false);
    pa0 += g.kcsA; pa1 += g.kcsA; pb0 += 32; pb1 += 32;
  }

  const long long cB = blockIdx.y * g.sC1 + blockIdx.z * g.sC2;
  for (int j = 0; j < 2; ++j) {
    const int bn = tn * 32 + j * 16 + nl;
    if (bn >= g.N) continue;
    const float biasv = g.bias ? g.bias[bn] : 0.f;
    for (int i = 0; i < 2; ++i) {
      const v8f acc = i == 0 ? (j == 0 ? a00 : a01) : (j == 0 ? a10 : a11);
      for (int r = 0; r < 8; ++r) {
        const int cm = tm * 32 + i * 16 + r + 8 * grp;   // C layout: M = reg + 8*grp
        if (cm >= g.M) break;
        float v = acc[r] + biasv;
        if (g.flags & FLAG_RELU) v = fmaxf(v, 0.f);
        if (g.flags & FLAG_ROWSCALE) v *= g.rowscale[cm];
        const long long rel = (long long)cm * g.ldc + (long long)bn * g.colStrideC;
        if (g.flags & FLAG_GATE) v *= 1.f / (1.f + __expf(-g.gate[rel]));
        const long long ci = cB + rel;
        if (g.flags & FLAG_F16OUT) {
          ((_Float16*)g.C)[ci] = (_Float16)v;
        } else {
          float* cp = (float*)g.C + ci;
          if (g.flags & FLAG_ACC) v += *cp;
          *cp = v;
        }
      }
    }
  }
}

// ---------------- LayerNorm (f32 in -> f16 out), blockDim == C --------------
__global__ void k_ln(const float* __restrict__ x, _Float16* __restrict__ y,
                     const float* __restrict__ sc, const float* __restrict__ bi, int C) {
  __shared__ float red[256];
  const long long r = blockIdx.x; const int t = threadIdx.x;
  float v = x[r * C + t];
  red[t] = v; __syncthreads();
  for (int s = C >> 1; s > 0; s >>= 1) { if (t < s) red[t] += red[t + s]; __syncthreads(); }
  const float mu = red[0] / C; __syncthreads();
  const float d = v - mu; red[t] = d * d; __syncthreads();
  for (int s = C >> 1; s > 0; s >>= 1) { if (t < s) red[t] += red[t + s]; __syncthreads(); }
  const float inv = rsqrtf(red[0] / C + 1e-5f);
  y[r * C + t] = (_Float16)(d * inv * sc[t] + bi[t]);
}

// ---------------- softmax over keys, fused scale + bias + mask --------------
__global__ void k_softmax(const float* __restrict__ Sc, _Float16* __restrict__ P,
                          const float* __restrict__ bias, int Hb,
                          const float* __restrict__ mask, long long mbs, long long mes,
                          int b0, int L, float scale) {
  __shared__ float red[256];
  const int q = blockIdx.x, bi = blockIdx.y, t = threadIdx.x;
  const int bg = b0 + bi, h = bg % Hb;
  const long long bb = bg / Hb;
  const long long base = ((long long)bi * L + q) * L;
  float v = Sc[base + t] * scale;
  if (bias) v += bias[((long long)q * L + t) * Hb + h];
  v += 1e9f * (mask[bb * mbs + (long long)t * mes] - 1.f);
  red[t] = v; __syncthreads();
  for (int s = L >> 1; s > 0; s >>= 1) { if (t < s) red[t] = fmaxf(red[t], red[t + s]); __syncthreads(); }
  const float mx = red[0]; __syncthreads();
  const float e = __expf(v - mx);
  red[t] = e; __syncthreads();
  for (int s = L >> 1; s > 0; s >>= 1) { if (t < s) red[t] += red[t + s]; __syncthreads(); }
  P[base + t] = (_Float16)(e / red[0]);
}

// ---------------- small elementwise / transpose helpers ---------------------
__global__ void k_cvt_transpose(const float* __restrict__ in, _Float16* __restrict__ out,
                                int R, int C) {
  const long long i = (long long)blockIdx.x * 256 + threadIdx.x;
  if (i >= (long long)R * C) return;
  const int c = (int)(i % C); const int r = (int)(i / C);
  out[(long long)c * R + r] = (_Float16)in[i];
}
__global__ void k_transpose_f16(const _Float16* __restrict__ in, _Float16* __restrict__ out,
                                int R, int C) {
  const long long i = (long long)blockIdx.x * 256 + threadIdx.x;
  if (i >= (long long)R * C) return;
  const int c = (int)(i % C); const int r = (int)(i / C);
  out[(long long)c * R + r] = in[i];
}
// v[b][l][h*ch+c] (strided) -> vt[((b*H+h)*ch+c)*L + l]
__global__ void k_make_vt(const _Float16* __restrict__ v, _Float16* __restrict__ vt,
                          int B, int L, int H, int ch, long long sB, long long sL) {
  long long i = (long long)blockIdx.x * 256 + threadIdx.x;
  if (i >= (long long)B * L * H * ch) return;
  const int c = (int)(i % ch); long long t = i / ch;
  const int h = (int)(t % H); t /= H;
  const int l = (int)(t % L); const int b = (int)(t / L);
  vt[(((long long)b * H + h) * ch + c) * L + l] =
      v[(long long)b * sB + (long long)l * sL + h * ch + c];
}
// out[(c*R1+r1)*R2+r2] = in[(r1*R2+r2)*C + c]  (or r1<->r2 swapped)
__global__ void k_perm(const _Float16* __restrict__ in, _Float16* __restrict__ out,
                       int R1, int R2, int C, int swap) {
  long long i = (long long)blockIdx.x * 256 + threadIdx.x;
  if (i >= (long long)R1 * R2 * C) return;
  const int c = (int)(i % C); long long rr = i / C;
  const int r2 = (int)(rr % R2); const int r1 = (int)(rr / R2);
  const long long src = swap ? ((long long)r2 * R1 + r1) * C + c
                             : ((long long)r1 * R2 + r2) * C + c;
  out[((long long)c * R1 + r1) * R2 + r2] = in[src];
}
// z[j*N+i][c] += d[i*N+j][c]  (transpose-accumulate for tri_att_end)
__global__ void k_add_T(float* __restrict__ z, const float* __restrict__ d, int N, int C) {
  long long i = (long long)blockIdx.x * 256 + threadIdx.x;
  if (i >= (long long)N * N * C) return;
  const int c = (int)(i % C); long long rr = i / C;
  const int jp = (int)(rr % N); const int ip = (int)(rr / N);
  z[((long long)jp * N + ip) * C + c] += d[i];
}
// out = sigmoid(g)*x  (gated attention output)
__global__ void k_gate(_Float16* __restrict__ out, const _Float16* __restrict__ g,
                       const float* __restrict__ x, long long n) {
  long long i = (long long)blockIdx.x * 256 + threadIdx.x;
  if (i >= n) return;
  const float gv = (float)g[i];
  out[i] = (_Float16)(x[i] / (1.f + __expf(-gv)));
}
// out = sigmoid(g)*p*rowmask  (triangle-mult a/b)
__global__ void k_gate_mask(_Float16* __restrict__ out, const _Float16* __restrict__ g,
                            const _Float16* __restrict__ p, const float* __restrict__ rm,
                            int cols, long long n) {
  long long i = (long long)blockIdx.x * 256 + threadIdx.x;
  if (i >= n) return;
  const float gv = (float)g[i];
  out[i] = (_Float16)((float)p[i] * rm[i / cols] / (1.f + __expf(-gv)));
}
// normrecip[i*N+j] = 1/(sum_s mask[s,i]*mask[s,j] + 1e-3)
__global__ void k_opm_norm(const float* __restrict__ mask, float* __restrict__ out,
                           int S, int N) {
  const int idx = blockIdx.x * 256 + threadIdx.x;
  if (idx >= N * N) return;
  const int i = idx / N, j = idx % N;
  float s = 0.f;
  for (int ss = 0; ss < S; ++ss) s += mask[ss * N + i] * mask[ss * N + j];
  out[idx] = 1.f / (s + 1e-3f);
}

// ---------------- host helpers ----------------------------------------------
static inline GemmArgs mkg(const _Float16* A, const _Float16* B, void* C,
                           int M, int N, int K,
                           long long lda, long long ldb, long long ldc,
                           int flags, const float* bias) {
  GemmArgs g;
  g.A = A; g.B = B; g.C = C; g.bias = bias; g.rowscale = nullptr; g.gate = nullptr;
  g.sA1 = g.sA2 = g.sB1 = g.sB2 = g.sC1 = g.sC2 = 0;
  g.rowDivA = 0x40000000; g.rsA1 = 0; g.rsA2 = lda; g.kcsA = 32;
  g.ldb = ldb; g.ldc = ldc; g.colStrideC = 1;
  g.M = M; g.N = N; g.K = K; g.flags = flags;
  return g;
}
static inline void launch_gemm(hipStream_t st, const GemmArgs& g, int B1, int B2) {
  const int tiles = ((g.M + 31) / 32) * ((g.N + 31) / 32);
  dim3 grid((tiles + 3) / 4, B1, B2);
  k_gemm<<<grid, 128, 0, st>>>(g);
}
static inline int blks(long long n) { return (int)((n + 255) / 256); }

// ============================================================================
extern "C" void kernel_launch(void* const* d_in, const int* in_sizes, int n_in,
                              void* d_out, int out_size, void* d_ws, size_t ws_size,
                              hipStream_t stream) {
  (void)in_sizes; (void)n_in; (void)out_size; (void)ws_size;
  char* ws = (char*)d_ws;
  auto F = [&](int i) { return (const float*)d_in[i]; };

  float* out_m = (float*)d_out;
  float* out_z = out_m + (size_t)S_ * N_ * CM;

  // ---- workspace layout (total 182 MiB) ----
  const size_t MB = 1ull << 20;
  _Float16* bufA  = (_Float16*)(ws + 4   * MB);   // 16 MiB: LN output
  _Float16* bufQ  = (_Float16*)(ws + 20  * MB);   // 16 MiB
  _Float16* bufK  = (_Float16*)(ws + 36  * MB);   // 16 MiB
  _Float16* bufV  = (_Float16*)(ws + 52  * MB);   // 16 MiB
  _Float16* bufG  = (_Float16*)(ws + 68  * MB);   // 16 MiB
  _Float16* bufVT = (_Float16*)(ws + 84  * MB);   // 16 MiB
  _Float16* bufP  = (_Float16*)(ws + 100 * MB);   // 16 MiB: probs / hidden / gated
  float*    bufS  = (float*)   (ws + 116 * MB);   // 32 MiB: scores / x
  float*    bufO  = (float*)   (ws + 148 * MB);   // 32 MiB: attn out / gates
  float*    bufZB = (float*)   (ws + 180 * MB);   //  2 MiB: z_bias / tri_b / norm

  // residual accumulators live in d_out
  hipMemcpyAsync(out_m, F(IN_M), (size_t)S_ * N_ * CM * 4, hipMemcpyDeviceToDevice, stream);
  hipMemcpyAsync(out_z, F(IN_Z), (size_t)N_ * N_ * CZ * 4, hipMemcpyDeviceToDevice, stream);

  // ---- pre-transpose all weights to f16 [cout][cin] (B-transposed GEMM op) --
  size_t wtOff = 0;
  auto wt = [&](int idx, int R, int C) -> const _Float16* {
    _Float16* dst = (_Float16*)(ws + wtOff);
    const int n = R * C;
    k_cvt_transpose<<<blks(n), 256, 0, stream>>>(F(idx), dst, R, C);
    wtOff += ((size_t)n * 2 + 255) & ~(size_t)255;
    return dst;
  };
  const _Float16 *ra_q = wt(RA_Q_W, CM, CM), *ra_k = wt(RA_K_W, CM, CM),
                 *ra_v = wt(RA_V_W, CM, CM), *ra_g = wt(RA_G_W, CM, CM),
                 *ra_o = wt(RA_O_W, CM, CM), *ra_zb = wt(RA_ZB_W, CZ, HM);
  const _Float16 *ca_q = wt(CA_Q_W, CM, CM), *ca_k = wt(CA_K_W, CM, CM),
                 *ca_v = wt(CA_V_W, CM, CM), *ca_g = wt(CA_G_W, CM, CM),
                 *ca_o = wt(CA_O_W, CM, CM);
  const _Float16 *mt1 = wt(MT_L1_W, CM, TN * CM), *mt2 = wt(MT_L2_W, TN * CM, CM);
  const _Float16 *op1 = wt(OPM_L1_W, CM, COPM), *op2 = wt(OPM_L2_W, CM, COPM),
                 *opo = wt(OPM_OUT_W, COPM * COPM, CZ);
  const _Float16 *pt1 = wt(PT_L1_W, CZ, TN * CZ), *pt2 = wt(PT_L2_W, TN * CZ, CZ);
  const _Float16 *to_ag = wt(TMO_AG_W, CZ, CMUL), *to_ap = wt(TMO_AP_W, CZ, CMUL),
                 *to_bg = wt(TMO_BG_W, CZ, CMUL), *to_bp = wt(TMO_BP_W, CZ, CMUL),
                 *to_g  = wt(TMO_G_W, CZ, CZ),    *to_o  = wt(TMO_OUT_W, CMUL, CZ);
  const _Float16 *ti_ag = wt(TMI_AG_W, CZ, CMUL), *ti_ap = wt(TMI_AP_W, CZ, CMUL),
                 *ti_bg = wt(TMI_BG_W, CZ, CMUL), *ti_bp = wt(TMI_BP_W, CZ, CMUL),
                 *ti_g  = wt(TMI_G_W, CZ, CZ),    *ti_o  = wt(TMI_OUT_W, CMUL, CZ);
  const _Float16 *ts_q = wt(TAS_Q_W, CZ, CZ), *ts_k = wt(TAS_K_W, CZ, CZ),
                 *ts_v = wt(TAS_V_W, CZ, CZ), *ts_g = wt(TAS_G_W, CZ, CZ),
                 *ts_o = wt(TAS_O_W, CZ, CZ), *ts_tb = wt(TAS_TB_W, CZ, HP);
  const _Float16 *te_q = wt(TAE_Q_W, CZ, CZ), *te_k = wt(TAE_K_W, CZ, CZ),
                 *te_v = wt(TAE_V_W, CZ, CZ), *te_g = wt(TAE_G_W, CZ, CZ),
                 *te_o = wt(TAE_O_W, CZ, CZ), *te_tb = wt(TAE_TB_W, CZ, HP);

  // q/k/v/g projections into bufQ/K/V/G (optionally with A-row remap)
  auto proj4 = [&](const _Float16* ln, int M, int K,
                   const _Float16* wq, const float* bq, const _Float16* wk, const float* bk,
                   const _Float16* wv, const float* bv, const _Float16* wg, const float* bg2,
                   int HC, int rowDiv, long long rs1, long long rs2) {
    const _Float16* W4[4] = { wq, wk, wv, wg };
    const float*    B4[4] = { bq, bk, bv, bg2 };
    _Float16*       O4[4] = { bufQ, bufK, bufV, bufG };
    for (int i = 0; i < 4; ++i) {
      GemmArgs g = mkg(ln, W4[i], O4[i], M, HC, K, K, K, HC, FLAG_F16OUT, B4[i]);
      if (rowDiv) { g.rowDivA = rowDiv; g.rsA1 = rs1; g.rsA2 = rs2; }
      launch_gemm(stream, g, 1, 1);
    }
  };

  // generic attention middle: vt, chunked scores->softmax->P.V, gate -> bufP
  auto attention_core = [&](int B, int L, int H, int ch,
                            long long sB, long long sL, int chunkB,
                            const float* bias, const float* mask,
                            long long mbs, long long mes) {
    const long long tot = (long long)B * L * H * ch;
    k_make_vt<<<blks(tot), 256, 0, stream>>>(bufV, bufVT, B, L, H, ch, sB, sL);
    const float scale = 1.f / sqrtf((float)ch);
    for (int b0 = 0; b0 < B; b0 += chunkB) {
      GemmArgs gs = mkg(bufQ + (long long)b0 * sB, bufK + (long long)b0 * sB, bufS,
                        L, L, ch, sL, sL, L, 0, nullptr);
      gs.sA1 = sB; gs.sA2 = ch; gs.sB1 = sB; gs.sB2 = ch;
      gs.sC1 = (long long)H * L * L; gs.sC2 = (long long)L * L;
      launch_gemm(stream, gs, chunkB, H);
      k_softmax<<<dim3(L, chunkB * H), L, 0, stream>>>(bufS, bufP, bias, H,
                                                       mask, mbs, mes, b0 * H, L, scale);
      GemmArgs go = mkg(bufP, bufVT + (long long)b0 * H * ch * L,
                        bufO + (long long)b0 * sB, L, ch, L, L, L, sL, 0, nullptr);
      go.sA1 = (long long)H * L * L; go.sA2 = (long long)L * L;
      go.sB1 = (long long)H * ch * L; go.sB2 = (long long)ch * L;
      go.sC1 = sB; go.sC2 = ch;
      launch_gemm(stream, go, chunkB, H);
    }
    k_gate<<<blks(tot), 256, 0, stream>>>(bufP, bufG, bufO, tot);
  };

  // ======================= 1. MSA row attention (pair bias) ==================
  k_ln<<<S_ * N_, CM, 0, stream>>>(out_m, bufA, F(RA_LNM_SCALE), F(RA_LNM_BIAS), CM);
  k_ln<<<N_ * N_, CZ, 0, stream>>>(out_z, bufP, F(RA_LNZ_SCALE), F(RA_LNZ_BIAS), CZ);
  { GemmArgs g = mkg(bufP, ra_zb, bufZB, N_ * N_, HM, CZ, CZ, CZ, HM, 0, nullptr);
    launch_gemm(stream, g, 1, 1); }                        // z_bias [N,N,8]
  proj4(bufA, S_ * N_, CM, ra_q, F(RA_Q_B), ra_k, F(RA_K_B), ra_v, F(RA_V_B),
        ra_g, F(RA_G_B), HM * CHD, 0, 0, 0);
  attention_core(S_, N_, HM, CHD, (long long)N_ * CM, CM, 16,
                 bufZB, F(IN_MSA_MASK), N_, 1);
  { GemmArgs g = mkg(bufP, ra_o, out_m, S_ * N_, CM, CM, CM, CM, CM, FLAG_ACC, F(RA_O_B));
    launch_gemm(stream, g, 1, 1); }

  // ======================= 2. MSA column attention ===========================
  k_ln<<<S_ * N_, CM, 0, stream>>>(out_m, bufA, F(CA_LN_SCALE), F(CA_LN_BIAS), CM);
  proj4(bufA, S_ * N_, CM, ca_q, F(CA_Q_B), ca_k, F(CA_K_B), ca_v, F(CA_V_B),
        ca_g, F(CA_G_B), HM * CHD, 0, 0, 0);
  attention_core(N_, S_, HM, CHD, CM, (long long)N_ * CM, 32,
                 nullptr, F(IN_MSA_MASK), 1, N_);
  { GemmArgs g = mkg(bufP, ca_o, out_m, S_ * N_, CM, CM, CM, CM, CM, FLAG_ACC, F(CA_O_B));
    launch_gemm(stream, g, 1, 1); }

  // ======================= 3. MSA transition =================================
  k_ln<<<S_ * N_, CM, 0, stream>>>(out_m, bufA, F(MT_LN_SCALE), F(MT_LN_BIAS), CM);
  for (int r0 = 0; r0 < S_ * N_; r0 += 8192) {
    { GemmArgs g = mkg(bufA + (long long)r0 * CM, mt1, bufP, 8192, TN * CM, CM,
                       CM, CM, TN * CM, FLAG_RELU | FLAG_F16OUT, F(MT_L1_B));
      launch_gemm(stream, g, 1, 1); }
    { GemmArgs g = mkg(bufP, mt2, out_m + (long long)r0 * CM, 8192, CM, TN * CM,
                       TN * CM, TN * CM, CM, FLAG_ACC | FLAG_ROWSCALE, F(MT_L2_B));
      g.rowscale = F(IN_MSA_MASK) + r0;
      launch_gemm(stream, g, 1, 1); }
  }

  // ======================= 4. Outer product mean =============================
  k_ln<<<S_ * N_, CM, 0, stream>>>(out_m, bufA, F(OPM_LN_SCALE), F(OPM_LN_BIAS), CM);
  { GemmArgs g = mkg(bufA, op1, bufQ, S_ * N_, COPM, CM, CM, CM, COPM,
                     FLAG_F16OUT | FLAG_ROWSCALE, F(OPM_L1_B));
    g.rowscale = F(IN_MSA_MASK); launch_gemm(stream, g, 1, 1); }
  { GemmArgs g = mkg(bufA, op2, bufK, S_ * N_, COPM, CM, CM, CM, COPM,
                     FLAG_F16OUT | FLAG_ROWSCALE, F(OPM_L2_B));
    g.rowscale = F(IN_MSA_MASK); launch_gemm(stream, g, 1, 1); }
  { const long long n = (long long)S_ * N_ * COPM;
    k_transpose_f16<<<blks(n), 256, 0, stream>>>(bufQ, bufV, S_, N_ * COPM);  // a^T
    k_transpose_f16<<<blks(n), 256, 0, stream>>>(bufK, bufG, S_, N_ * COPM); } // b^T
  k_opm_norm<<<blks(N_ * N_), 256, 0, stream>>>(F(IN_MSA_MASK), bufZB, S_, N_);
  for (int i0 = 0; i0 < N_; i0 += 32) {
    // outer chunk [(i_loc,c) x (j,d)] = a^T[i,c,:] . b^T[j,d,:]
    { GemmArgs g = mkg(bufV + (long long)i0 * COPM * S_, bufG, bufP,
                       32 * COPM, N_ * COPM, S_, S_, S_, (long long)N_ * COPM,
                       FLAG_F16OUT, nullptr);
      launch_gemm(stream, g, 1, 1); }
    // (outer . Wout + b) / norm, += z   (A rows (i_loc,j), K-chunks = c)
    { GemmArgs g = mkg(bufP, opo, out_z + (long long)i0 * N_ * CZ,
                       32 * N_, CZ, COPM * COPM, 0, COPM * COPM, CZ,
                       FLAG_ACC | FLAG_ROWSCALE, F(OPM_OUT_B));
      g.rowDivA = N_; g.rsA1 = (long long)COPM * (N_ * COPM); g.rsA2 = COPM;
      g.kcsA = (long long)N_ * COPM;
      g.rowscale = bufZB + (long long)i0 * N_;
      launch_gemm(stream, g, 1, 1); }
  }

  // ======================= 5/6. Triangle multiplications =====================
  auto trimul = [&](const _Float16* wag, const _Float16* wap, const _Float16* wbg,
                    const _Float16* wbp, const _Float16* wg2, const _Float16* wo2,
                    const float* bag, const float* bap, const float* bbg, const float* bbp,
                    const float* bg2, const float* bo2,
                    const float* lni_s, const float* lni_b,
                    const float* lno_s, const float* lno_b, int swap) {
    k_ln<<<N_ * N_, CZ, 0, stream>>>(out_z, bufA, lni_s, lni_b, CZ);
    const long long tot = (long long)N_ * N_ * CMUL;
    { GemmArgs g = mkg(bufA, wag, bufQ, N_ * N_, CMUL, CZ, CZ, CZ, CMUL, FLAG_F16OUT, bag);
      launch_gemm(stream, g, 1, 1); }
    { GemmArgs g = mkg(bufA, wap, bufK, N_ * N_, CMUL, CZ, CZ, CZ, CMUL, FLAG_F16OUT, bap);
      launch_gemm(stream, g, 1, 1); }
    k_gate_mask<<<blks(tot), 256, 0, stream>>>(bufV, bufQ, bufK, F(IN_PAIR_MASK), CMUL, tot);
    { GemmArgs g = mkg(bufA, wbg, bufQ, N_ * N_, CMUL, CZ, CZ, CZ, CMUL, FLAG_F16OUT, bbg);
      launch_gemm(stream, g, 1, 1); }
    { GemmArgs g = mkg(bufA, wbp, bufK, N_ * N_, CMUL, CZ, CZ, CZ, CMUL, FLAG_F16OUT, bbp);
      launch_gemm(stream, g, 1, 1); }
    k_gate_mask<<<blks(tot), 256, 0, stream>>>(bufVT, bufQ, bufK, F(IN_PAIR_MASK), CMUL, tot);
    { GemmArgs g = mkg(bufA, wg2, bufO, N_ * N_, CZ, CZ, CZ, CZ, CZ, 0, bg2);
      launch_gemm(stream, g, 1, 1); }                    // gate logits (f32)
    k_perm<<<blks(tot), 256, 0, stream>>>(bufV,  bufQ, N_, N_, CMUL, swap);  // a_t[c][i][k]
    k_perm<<<blks(tot), 256, 0, stream>>>(bufVT, bufK, N_, N_, CMUL, swap);  // b_t[c][j][k]
    { GemmArgs g = mkg(bufQ, bufK, bufS, N_, N_, N_, N_, N_,
                       (long long)N_ * CMUL, 0, nullptr);  // x[i][j][c], batched over c
      g.colStrideC = CMUL; g.sA1 = (long long)N_ * N_; g.sB1 = (long long)N_ * N_;
      g.sC1 = 1;
      launch_gemm(stream, g, CMUL, 1); }
    k_ln<<<N_ * N_, CMUL, 0, stream>>>(bufS, bufP, lno_s, lno_b, CMUL);
    { GemmArgs g = mkg(bufP, wo2, out_z, N_ * N_, CZ, CMUL, CMUL, CMUL, CZ,
                       FLAG_ACC | FLAG_GATE, bo2);
      g.gate = bufO;
      launch_gemm(stream, g, 1, 1); }
  };
  trimul(to_ag, to_ap, to_bg, to_bp, to_g, to_o,
         F(TMO_AG_B), F(TMO_AP_B), F(TMO_BG_B), F(TMO_BP_B), F(TMO_G_B), F(TMO_OUT_B),
         F(TMO_LNI_SCALE), F(TMO_LNI_BIAS), F(TMO_LNO_SCALE), F(TMO_LNO_BIAS), 0);
  trimul(ti_ag, ti_ap, ti_bg, ti_bp, ti_g, ti_o,
         F(TMI_AG_B), F(TMI_AP_B), F(TMI_BG_B), F(TMI_BP_B), F(TMI_G_B), F(TMI_OUT_B),
         F(TMI_LNI_SCALE), F(TMI_LNI_BIAS), F(TMI_LNO_SCALE), F(TMI_LNO_BIAS), 1);

  // ======================= 7. Triangle attention (start) =====================
  k_ln<<<N_ * N_, CZ, 0, stream>>>(out_z, bufA, F(TAS_LN_SCALE), F(TAS_LN_BIAS), CZ);
  { GemmArgs g = mkg(bufA, ts_tb, bufZB, N_ * N_, HP, CZ, CZ, CZ, HP, 0, nullptr);
    launch_gemm(stream, g, 1, 1); }
  proj4(bufA, N_ * N_, CZ, ts_q, F(TAS_Q_B), ts_k, F(TAS_K_B), ts_v, F(TAS_V_B),
        ts_g, F(TAS_G_B), CZ, 0, 0, 0);
  attention_core(N_, N_, HP, CHD, (long long)N_ * CZ, CZ, 32,
                 bufZB, F(IN_PAIR_MASK), N_, 1);
  { GemmArgs g = mkg(bufP, ts_o, out_z, N_ * N_, CZ, CZ, CZ, CZ, CZ, FLAG_ACC, F(TAS_O_B));
    launch_gemm(stream, g, 1, 1); }

  // ======================= 8. Triangle attention (end) =======================
  // Work in transposed (j,i) space by remapping A rows of every LN consumer.
  k_ln<<<N_ * N_, CZ, 0, stream>>>(out_z, bufA, F(TAE_LN_SCALE), F(TAE_LN_BIAS), CZ);
  { GemmArgs g = mkg(bufA, te_tb, bufZB, N_ * N_, HP, CZ, 0, CZ, HP, 0, nullptr);
    g.rowDivA = N_; g.rsA1 = CZ; g.rsA2 = (long long)N_ * CZ;
    launch_gemm(stream, g, 1, 1); }
  proj4(bufA, N_ * N_, CZ, te_q, F(TAE_Q_B), te_k, F(TAE_K_B), te_v, F(TAE_V_B),
        te_g, F(TAE_G_B), CZ, N_, CZ, (long long)N_ * CZ);
  attention_core(N_, N_, HP, CHD, (long long)N_ * CZ, CZ, 32,
                 bufZB, F(IN_PAIR_MASK), 1, N_);
  { GemmArgs g = mkg(bufP, te_o, bufO, N_ * N_, CZ, CZ, CZ, CZ, CZ, 0, F(TAE_O_B));
    launch_gemm(stream, g, 1, 1); }
  { const long long tot = (long long)N_ * N_ * CZ;
    k_add_T<<<blks(tot), 256, 0, stream>>>(out_z, bufO, N_, CZ); }

  // ======================= 9. Pair transition ================================
  k_ln<<<N_ * N_, CZ, 0, stream>>>(out_z, bufA, F(PT_LN_SCALE), F(PT_LN_BIAS), CZ);
  for (int r0 = 0; r0 < N_ * N_; r0 += 16384) {
    { GemmArgs g = mkg(bufA + (long long)r0 * CZ, pt1, bufP, 16384, TN * CZ, CZ,
                       CZ, CZ, TN * CZ, FLAG_RELU | FLAG_F16OUT, F(PT_L1_B));
      launch_gemm(stream, g, 1, 1); }
    { GemmArgs g = mkg(bufP, pt2, out_z + (long long)r0 * CZ, 16384, CZ, TN * CZ,
                       TN * CZ, TN * CZ, CZ, FLAG_ACC | FLAG_ROWSCALE, F(PT_L2_B));
      g.rowscale = F(IN_PAIR_MASK) + r0;
      launch_gemm(stream, g, 1, 1); }
  }
}
